// ReAttentionMSA_75960791597710
// MI455X (gfx1250) — matrix operand order repaired and checked by reading, not compiled, not run on hardware
//
#include <hip/hip_runtime.h>
#include <math.h>

typedef _Float16 f16;
typedef __attribute__((ext_vector_type(16))) _Float16 v16h;
typedef __attribute__((ext_vector_type(8)))  _Float16 v8h;
typedef __attribute__((ext_vector_type(8)))  float    v8f;
typedef __attribute__((ext_vector_type(4)))  float    v4f;

#define NB 8
#define NT 1024
#define NE 384
#define NH 12
#define ND 32
#define NPAIR 78            // 12*13/2 lower-tri incl diag
#define BN_EPS 1e-5f
#define PSTRIDE 36          // f32 prob row stride (144B: 16B-aligned, conflict-free over t)
#define PH 40               // f16 prob row stride (80B: 16B-aligned, conflict-free over t)

__device__ __forceinline__ v8f wmma_f16(v16h a, v16h b, v8f c) {
  // v_wmma_f32_16x16x32_f16: D = A(16x32 f16) * B(32x16 f16) + C(16x16 f32)
  return __builtin_amdgcn_wmma_f32_16x16x32_f16(false, a, false, b, (short)0, c, false, false);
}

// A-matrix 16x32 f16 fragment from row-major memory.
// lane<16: row=lane,  K = 0..15 ; lane>=16: row=lane-16, K = 16..31
__device__ __forceinline__ v16h load_a_frag(const f16* base, int rowStrideH, int lane) {
  const f16* r = base + (size_t)(lane & 15) * rowStrideH + ((lane & 16) ? 16 : 0);
  v8h lo = *(const v8h*)(r);
  v8h hi = *(const v8h*)(r + 8);
  return __builtin_shufflevector(lo, hi, 0,1,2,3,4,5,6,7,8,9,10,11,12,13,14,15);
}

// B-matrix 32x16 fragment, element (k, n) = Mrowmajor[n][k] (k contiguous, 32/row):
// lanes 0-15: N=lane, K=0..15 ; lanes 16-31: N=lane-16, K=16..31 -> one 32B load/lane
__device__ __forceinline__ v16h load_b_rows(const f16* rows, int lane) {
  return *(const v16h*)(rows + (size_t)(lane & 15) * ND + ((lane & 16) ? 16 : 0));
}

// B-matrix 32x16 fragment for ren@V: element (k=s, n=d) = Vt[d][s] (Vt is [D][T], s contig)
__device__ __forceinline__ v16h load_b_vt(const f16* vthead, int d0, int s0, int lane) {
  return *(const v16h*)(vthead + (size_t)(d0 + (lane & 15)) * NT + s0 + ((lane & 16) ? 16 : 0));
}

__device__ __forceinline__ float vmax8(v8f z) {
  float a0 = fmaxf(z[0], z[1]);
  float a1 = fmaxf(z[2], z[3]);
  float a2 = fmaxf(z[4], z[5]);
  float a3 = fmaxf(z[6], z[7]);
  return fmaxf(fmaxf(a0, a1), fmaxf(a2, a3));
}
__device__ __forceinline__ float expsum8(v8f z, float mn) {
  float s = 0.f;
#pragma unroll
  for (int r = 0; r < 8; ++r) s += __expf(z[r] - mn);
  return s;
}

// ---------------------------------------------------------------------------
// Kernel 1: per-head QKV projection. x[B,T,E] -> Q,K f16 [B,H,T,D] (scale folded
// into Q), V stored transposed f16 [B,H,D,T] for contiguous B-fragment loads.
// ---------------------------------------------------------------------------
__global__ __launch_bounds__(128) void k_qkv(const float* __restrict__ x,
                                             const float* __restrict__ Wqkv,
                                             const float* __restrict__ bqkv,
                                             f16* __restrict__ Q, f16* __restrict__ Kh,
                                             f16* __restrict__ Vt) {
  __shared__ float xs[NE];
  __shared__ float Ws[96 * 32];
  const int bt = blockIdx.x;            // 0..B*T-1
  const int b = bt >> 10, t = bt & (NT - 1);
  const int tid = threadIdx.x;
  for (int i = tid; i < NE; i += 128) xs[i] = x[(size_t)bt * NE + i];
  for (int i = tid; i < 96 * 32; i += 128) Ws[i] = Wqkv[i];
  __syncthreads();
  const float scale = 0.17677669529663687f;   // 1/sqrt(32)
  for (int o = tid; o < NH * 96; o += 128) {
    int h = o / 96, e = o % 96;
    const float* xr = xs + h * ND;
    const float* wr = Ws + e * ND;
    float acc = bqkv[e];
#pragma unroll
    for (int d = 0; d < ND; ++d) acc = fmaf(xr[d], wr[d], acc);
    const size_t bh = (size_t)b * NH + h;
    if (e < 32) {
      Q[(bh * NT + t) * ND + e] = (f16)(acc * scale);
    } else if (e < 64) {
      Kh[(bh * NT + t) * ND + (e - 32)] = (f16)acc;
    } else {
      Vt[(bh * ND + (e - 64)) * NT + t] = (f16)acc;
    }
  }
}

// ---------------------------------------------------------------------------
// Kernel 1b: Vsum[b,h,d] = sum_s V (from the f16 the GEMM actually consumes)
// ---------------------------------------------------------------------------
__global__ __launch_bounds__(128) void k_vsum(const f16* __restrict__ Vt,
                                              float* __restrict__ Vsum) {
  const int bh = blockIdx.x;
  const int tid = threadIdx.x;
  const int d = tid >> 2, c = tid & 3;
  const f16* p = Vt + ((size_t)bh * ND + d) * NT + c * 256;
  float s = 0.f;
  for (int i = 0; i < 256; ++i) s += (float)p[i];
  __shared__ float red[128];
  red[tid] = s;
  __syncthreads();
  if (c == 0) Vsum[bh * ND + d] = red[tid] + red[tid + 1] + red[tid + 2] + red[tid + 3];
}

// ---------------------------------------------------------------------------
// Kernel 2: softmax row stats + 12x12 cross second-moment. 384 thr = 12 waves,
// wave w owns head w, t-tile = 32. Scores computed TRANSPOSED (z^T = K Q^T) so
// the softmax axis lives on VGPRs: in-lane reductions + one shfl_xor(16).
// ---------------------------------------------------------------------------
__global__ __launch_bounds__(384) void k_stats(const f16* __restrict__ Q,
                                               const f16* __restrict__ Kg,
                                               float* __restrict__ Mrow,
                                               float* __restrict__ Lrow,
                                               float* __restrict__ Mmat) {
  __shared__ __align__(16) float pLDS[NH * 32 * PSTRIDE];
  __shared__ float Mred[NPAIR];
  const int b = blockIdx.y;
  const int t0 = blockIdx.x * 32;
  const int tid = threadIdx.x;
  const int lane = tid & 31;
  const int w = tid >> 5;                // wave id == head id
  const int hf = (lane >> 4) & 1;
  if (tid < NPAIR) Mred[tid] = 0.f;

  const size_t bh = (size_t)b * NH + w;
  const f16* Qh = Q + (bh * NT + t0) * ND;
  const f16* Kh = Kg + bh * NT * ND;

  // B-fragments of Q (columns t of z^T); A-fragments of K stream per s-tile
  v16h qb[2];
  qb[0] = load_b_rows(Qh, lane);
  qb[1] = load_b_rows(Qh + 16 * ND, lane);

  const v8f zero8 = {0.f,0.f,0.f,0.f,0.f,0.f,0.f,0.f};
  float m[2] = {-1e30f, -1e30f};
  float l[2] = {0.f, 0.f};

  // ---- sweep 1: exact online max / denominator, per-lane column state ----
  for (int s0 = 0; s0 < NT; s0 += 32) {
    v16h ka0 = load_a_frag(Kh + (size_t)s0 * ND, ND, lane);
    v16h ka1 = load_a_frag(Kh + (size_t)(s0 + 16) * ND, ND, lane);
#pragma unroll
    for (int ct = 0; ct < 2; ++ct) {
      v8f z0 = wmma_f16(ka0, qb[ct], zero8);
      v8f z1 = wmma_f16(ka1, qb[ct], zero8);
      float tm = fmaxf(vmax8(z0), vmax8(z1));
      tm = fmaxf(tm, __shfl_xor(tm, 16, 32));
      float mn = fmaxf(m[ct], tm);
      float es = expsum8(z0, mn) + expsum8(z1, mn);
      es += __shfl_xor(es, 16, 32);
      l[ct] = l[ct] * __expf(m[ct] - mn) + es;
      m[ct] = mn;
    }
  }
  float linv[2] = {1.0f / l[0], 1.0f / l[1]};
  if (lane < 16) {
#pragma unroll
    for (int ct = 0; ct < 2; ++ct) {
      size_t idx = bh * NT + t0 + ct * 16 + lane;
      Mrow[idx] = m[ct];
      Lrow[idx] = l[ct];
    }
  }

  // pair ownership: 78 pairs x 4 row-quarters = 312 threads
  int pi = 0, pj = 0;
  const int pairId = tid >> 2;
  const int quad = tid & 3;
  if (tid < 312) {
    int i = 0;
    while (((i + 1) * (i + 2)) / 2 <= pairId) ++i;
    pi = i;
    pj = pairId - (i * (i + 1)) / 2;
  }

  // ---- sweep 2: probabilities into LDS (b128 packed), accumulate M[h,h'] ----
  v4f macc4 = {0.f, 0.f, 0.f, 0.f};
  for (int s0 = 0; s0 < NT; s0 += 32) {
    v16h ka0 = load_a_frag(Kh + (size_t)s0 * ND, ND, lane);
    v16h ka1 = load_a_frag(Kh + (size_t)(s0 + 16) * ND, ND, lane);
#pragma unroll
    for (int ct = 0; ct < 2; ++ct) {
      v8f z0 = wmma_f16(ka0, qb[ct], zero8);
      v8f z1 = wmma_f16(ka1, qb[ct], zero8);
      const int tcol = w * 32 + ct * 16 + (lane & 15);
      float* dst = pLDS + tcol * PSTRIDE + 8 * hf;
      v4f p0a, p0b, p1a, p1b;
#pragma unroll
      for (int r = 0; r < 4; ++r) {
        p0a[r] = __expf(z0[r]     - m[ct]) * linv[ct];
        p0b[r] = __expf(z0[r + 4] - m[ct]) * linv[ct];
        p1a[r] = __expf(z1[r]     - m[ct]) * linv[ct];
        p1b[r] = __expf(z1[r + 4] - m[ct]) * linv[ct];
      }
      *(v4f*)(dst)      = p0a;
      *(v4f*)(dst + 4)  = p0b;
      *(v4f*)(dst + 16) = p1a;
      *(v4f*)(dst + 20) = p1b;
    }
    __syncthreads();
    if (tid < 312) {
      const v4f* Pi4 = (const v4f*)(pLDS + pi * 32 * PSTRIDE);
      const v4f* Pj4 = (const v4f*)(pLDS + pj * 32 * PSTRIDE);
      const int tl0 = quad * 8;
#pragma unroll
      for (int tl = 0; tl < 8; ++tl) {
        const int row4 = (tl0 + tl) * (PSTRIDE / 4);
#pragma unroll
        for (int sv = 0; sv < 8; ++sv) macc4 += Pi4[row4 + sv] * Pj4[row4 + sv];
      }
    }
    __syncthreads();
  }
  if (tid < 312) {
    float macc = macc4[0] + macc4[1] + macc4[2] + macc4[3];
    atomicAdd(&Mred[pairId], macc);
  }
  __syncthreads();
  if (tid < NPAIR) atomicAdd(&Mmat[tid], Mred[tid]);
}

// ---------------------------------------------------------------------------
// Kernel 3: BN coefficients from analytic mean + 12x12 second moment.
// ---------------------------------------------------------------------------
__global__ __launch_bounds__(64) void k_coef(const float* __restrict__ Wconv,
                                             const float* __restrict__ bconv,
                                             const float* __restrict__ gamma,
                                             const float* __restrict__ beta,
                                             const float* __restrict__ Mmat,
                                             float* __restrict__ W2,
                                             float* __restrict__ c2) {
  const int g = threadIdx.x;
  if (g >= NH) return;
  const float invN = 1.0f / ((float)NB * (float)NT * (float)NT);
  float ws = 0.f;
  for (int h = 0; h < NH; ++h) ws += Wconv[g * NH + h];
  const float mean = ws / (float)NT + bconv[g];
  float e2 = 0.f;
  for (int i = 0; i < NH; ++i)
    for (int j = 0; j < NH; ++j) {
      int a = i > j ? i : j, bb = i > j ? j : i;
      float Mij = Mmat[(a * (a + 1)) / 2 + bb] * invN;
      e2 += Wconv[g * NH + i] * Wconv[g * NH + j] * Mij;
    }
  e2 += 2.f * bconv[g] * ws / (float)NT + bconv[g] * bconv[g];
  const float var = e2 - mean * mean;
  const float ag = gamma[g] * rsqrtf(var + BN_EPS);
  const float cg = beta[g] - ag * mean;
  for (int h = 0; h < NH; ++h) W2[g * NH + h] = ag * Wconv[g * NH + h];
  c2[g] = ag * bconv[g] + cg;
}

// ---------------------------------------------------------------------------
// Kernel 4: recompute probabilities (transposed) into f16 LDS, cross-head mix
// in packed f16 directly into WMMA A-fragments, second WMMA vs V, + c2_g*Vsum.
// ---------------------------------------------------------------------------
__global__ __launch_bounds__(384) void k_out(const f16* __restrict__ Q,
                                             const f16* __restrict__ Kg,
                                             const f16* __restrict__ Vt,
                                             const float* __restrict__ Mrow,
                                             const float* __restrict__ Lrow,
                                             const float* __restrict__ W2,
                                             const float* __restrict__ c2v,
                                             const float* __restrict__ Vsum,
                                             float* __restrict__ out) {
  __shared__ __align__(16) f16 pLDS[NH * 32 * PH];     // 30 KB f16 staging
  const int b = blockIdx.y;
  const int t0 = blockIdx.x * 32;
  const int tid = threadIdx.x;
  const int lane = tid & 31;
  const int w = tid >> 5;                 // wave id == head id (h and g roles)
  const int hf = (lane >> 4) & 1;

  const size_t bh = (size_t)b * NH + w;
  const f16* Qh = Q + (bh * NT + t0) * ND;
  const f16* Kh = Kg + bh * NT * ND;
  const f16* Vh = Vt + bh * (size_t)ND * NT;

  f16 w2h[NH];
#pragma unroll
  for (int h = 0; h < NH; ++h) w2h[h] = (f16)W2[w * NH + h];
  const float c2g = c2v[w];

  v16h qb[2];
  qb[0] = load_b_rows(Qh, lane);
  qb[1] = load_b_rows(Qh + 16 * ND, lane);

  float m[2], linv[2];
#pragma unroll
  for (int ct = 0; ct < 2; ++ct) {
    size_t idx = bh * NT + t0 + ct * 16 + (lane & 15);
    m[ct] = Mrow[idx];
    linv[ct] = 1.0f / Lrow[idx];
  }

  const v8f zero8 = {0.f,0.f,0.f,0.f,0.f,0.f,0.f,0.f};
  const v8h zero8h = {(f16)0,(f16)0,(f16)0,(f16)0,(f16)0,(f16)0,(f16)0,(f16)0};
  v8f acc[2][2];                           // [rt][dt]
#pragma unroll
  for (int a0 = 0; a0 < 2; ++a0)
#pragma unroll
    for (int a1 = 0; a1 < 2; ++a1) acc[a0][a1] = zero8;

  const int sb = hf * 8;                   // this lane's contiguous s-chunk base

  for (int s0 = 0; s0 < NT; s0 += 32) {
    // probabilities of own head -> f16 LDS (2x b128 per z-pair)
    v16h ka0 = load_a_frag(Kh + (size_t)s0 * ND, ND, lane);
    v16h ka1 = load_a_frag(Kh + (size_t)(s0 + 16) * ND, ND, lane);
#pragma unroll
    for (int ct = 0; ct < 2; ++ct) {
      v8f z0 = wmma_f16(ka0, qb[ct], zero8);
      v8f z1 = wmma_f16(ka1, qb[ct], zero8);
      const int tcol = w * 32 + ct * 16 + (lane & 15);
      v8h ps0, ps1;
#pragma unroll
      for (int r = 0; r < 8; ++r) {
        ps0[r] = (f16)(__expf(z0[r] - m[ct]) * linv[ct]);
        ps1[r] = (f16)(__expf(z1[r] - m[ct]) * linv[ct]);
      }
      f16* dst = pLDS + tcol * PH + sb;
      *(v8h*)(dst)      = ps0;
      *(v8h*)(dst + 16) = ps1;
    }
    __syncthreads();
    // cross-head mix in packed f16; accumulators ARE the A-fragment halves
    v16h af[2];
#pragma unroll
    for (int rt = 0; rt < 2; ++rt) {
      const int trow = rt * 16 + (lane & 15);
      v8h aLo = zero8h, aHi = zero8h;
#pragma unroll
      for (int h = 0; h < NH; ++h) {
        const f16* Pr = pLDS + (h * 32 + trow) * PH + hf * 16;
        v8h lo = *(const v8h*)(Pr);
        v8h hi8 = *(const v8h*)(Pr + 8);
        const f16 wv = w2h[h];
        const v8h w8 = {wv, wv, wv, wv, wv, wv, wv, wv};
        aLo += w8 * lo;
        aHi += w8 * hi8;
      }
      af[rt] = __builtin_shufflevector(aLo, aHi, 0,1,2,3,4,5,6,7,8,9,10,11,12,13,14,15);
    }
    __syncthreads();
    // ren(32t x 32s) @ V(32s x 32d)
#pragma unroll
    for (int dt = 0; dt < 2; ++dt) {
      v16h bv = load_b_vt(Vh, dt * 16, s0, lane);
#pragma unroll
      for (int rt = 0; rt < 2; ++rt) {
        acc[rt][dt] = wmma_f16(af[rt], bv, acc[rt][dt]);
      }
    }
  }
  // epilogue: add c2_g * Vsum, scatter to [B,T,E]
#pragma unroll
  for (int dt = 0; dt < 2; ++dt) {
    float vs = Vsum[((size_t)b * NH + w) * ND + dt * 16 + (lane & 15)];
#pragma unroll
    for (int rt = 0; rt < 2; ++rt)
#pragma unroll
      for (int r = 0; r < 8; ++r) {
        int t = t0 + rt * 16 + r + 8 * hf;
        out[((size_t)b * NT + t) * NE + w * ND + dt * 16 + (lane & 15)] =
            acc[rt][dt][r] + c2g * vs;
      }
  }
}

// ---------------------------------------------------------------------------
extern "C" void kernel_launch(void* const* d_in, const int* in_sizes, int n_in,
                              void* d_out, int out_size, void* d_ws, size_t ws_size,
                              hipStream_t stream) {
  (void)in_sizes; (void)n_in; (void)out_size; (void)ws_size;
  const float* x     = (const float*)d_in[0];
  const float* Wqkv  = (const float*)d_in[1];
  const float* bqkv  = (const float*)d_in[2];
  const float* Wconv = (const float*)d_in[3];
  const float* bconv = (const float*)d_in[4];
  const float* gamma = (const float*)d_in[5];
  const float* beta  = (const float*)d_in[6];
  float* out = (float*)d_out;

  size_t off = 0;
  auto carve = [&](size_t bytes) -> void* {
    void* p = (void*)((char*)d_ws + off);
    off += (bytes + 255) & ~(size_t)255;
    return p;
  };
  f16* Q     = (f16*)carve((size_t)NB * NH * NT * ND * sizeof(f16));
  f16* Kd    = (f16*)carve((size_t)NB * NH * NT * ND * sizeof(f16));
  f16* Vt    = (f16*)carve((size_t)NB * NH * NT * ND * sizeof(f16));
  float* Mr  = (float*)carve((size_t)NB * NH * NT * sizeof(float));
  float* Lr  = (float*)carve((size_t)NB * NH * NT * sizeof(float));
  float* Mm  = (float*)carve(NPAIR * sizeof(float));
  float* W2  = (float*)carve(NH * NH * sizeof(float));
  float* c2  = (float*)carve(NH * sizeof(float));
  float* Vs  = (float*)carve((size_t)NB * NH * ND * sizeof(float));

  hipMemsetAsync(Mm, 0, NPAIR * sizeof(float), stream);
  k_qkv <<<NB * NT, 128, 0, stream>>>(x, Wqkv, bqkv, Q, Kd, Vt);
  k_vsum<<<NB * NH, 128, 0, stream>>>(Vt, Vs);
  k_stats<<<dim3(NT / 32, NB), 384, 0, stream>>>(Q, Kd, Mr, Lr, Mm);
  k_coef<<<1, 64, 0, stream>>>(Wconv, bconv, gamma, beta, Mm, W2, c2);
  k_out <<<dim3(NT / 32, NB), 384, 0, stream>>>(Q, Kd, Vt, Mr, Lr, W2, c2, Vs, out);
}
